// SplatGaussian2D_43550968382049
// MI455X (gfx1250) — matrix operand order repaired and checked by reading, not compiled, not run on hardware
//
#include <hip/hip_runtime.h>

typedef __attribute__((ext_vector_type(16))) _Float16 v16h;
typedef __attribute__((ext_vector_type(8)))  float    v8f;

#define NB      8192
#define NG      2048
#define CHUNK   64
#define PAIRS   (CHUNK/2)          // 32 gaussian pairs per chunk
#define NCHUNK  (NG/CHUNK)         // 32 chunks
#define RAWF    (CHUNK*27)         // raw f32 features per chunk
#define RAWB    (RAWF*4)           // 6912 bytes
#define NB128   (RAWB/16)          // 432 16-byte transfers
#define S_MIN   0.033333333f
#define S_SPAN  1.3f               // S_MAX - S_MIN = 4/3 - 1/30
#define L2E     1.4426950408889634f

__device__ __forceinline__ unsigned pkh(float a, float b) {
  auto h = __builtin_amdgcn_cvt_pkrtz(a, b);   // __fp16 ext_vector(2)
  return __builtin_bit_cast(unsigned, h);
}
__device__ __forceinline__ float sigm(float x) {
  float e = __builtin_amdgcn_exp2f(x * -L2E);
  return __builtin_amdgcn_rcpf(1.0f + e);
}
__device__ __forceinline__ void async_b128(unsigned lds_addr, unsigned voff,
                                           unsigned long long sbase) {
  // GLOBAL_LOAD_ASYNC_TO_LDS_B128 (GVS mode): LDS[vdst] = MEM[saddr + vaddr]
  asm volatile("global_load_async_to_lds_b128 %0, %1, %2"
               :: "v"(lds_addr), "v"(voff), "s"(sbase) : "memory");
}
__device__ __forceinline__ void wait_async0() {
  asm volatile("s_wait_asynccnt 0" ::: "memory");
}

__global__ __launch_bounds__(128)
void splat2d_kernel(const float* __restrict__ xb,    // [B,2]
                    const float* __restrict__ xyz,   // [N,2]
                    const float* __restrict__ feat,  // [N,9,3]
                    const float* __restrict__ opac,  // [N]
                    const float* __restrict__ scal,  // [N,2]
                    const float* __restrict__ rot,   // [N]
                    float* __restrict__ out)         // [B,3]
{
  __shared__ __align__(32) float    rawFeat[RAWF];        // async landing zone
  __shared__ __align__(32) float    params[CHUNK * 8];    // gx,gy,cos,sin,S0,S1,op,pad
  __shared__ __align__(32) unsigned tiles[PAIRS * 96 + 8];// f16 WMMA A-tiles + zero slot

  const int tid  = threadIdx.x;
  const int lane = tid & 31;
  const int col  = lane & 15;       // D column = ray-in-wave
  const int half = lane >> 4;       // which gaussian of the pair this lane owns
  const int ray  = blockIdx.x * 64 + (tid >> 5) * 16 + col;

  const float px = xb[2 * ray];
  const float py = xb[2 * ray + 1];

  if (tid < 8) tiles[PAIRS * 96 + tid] = 0u;      // shared zero A-row

  const unsigned ldsRaw   = (unsigned)(unsigned long long)(void*)&rawFeat[0];
  const unsigned zOff     = PAIRS * 96;           // uint index of zero slot
  const unsigned slotOff  = (unsigned)((col + 6 * half) * 8);

  float acc0 = 0.f, acc1 = 0.f, acc2 = 0.f;

  // prologue: async-stage chunk 0 raw features into LDS
  {
    unsigned long long sb = (unsigned long long)feat;
    #pragma unroll
    for (int i = 0; i < 4; ++i) {
      int idx = i * 128 + tid;
      if (idx < NB128) async_b128(ldsRaw + idx * 16, (unsigned)(idx * 16), sb);
    }
  }

  for (int ch = 0; ch < NCHUNK; ++ch) {
    wait_async0();
    __syncthreads();

    // ---- stage: convert features -> f16 A-tiles, precompute params ----
    if (tid < CHUNK) {                      // one thread per gaussian: A-tile
      const int j  = tid & 1;
      const int pb = (tid >> 1) * 96;
      const float* f = &rawFeat[tid * 27];
      #pragma unroll
      for (int cc = 0; cc < 3; ++cc) {
        unsigned q0 = pkh(f[0 + cc],  f[3 + cc]);
        unsigned q1 = pkh(f[6 + cc],  f[9 + cc]);
        unsigned q2 = pkh(f[12 + cc], f[15 + cc]);
        unsigned q3 = pkh(f[18 + cc], f[21 + cc]);
        unsigned q4 = pkh(f[24 + cc], 0.f);
        int lo = pb + (j * 3 + cc) * 8;       // rows for lanes 0..5
        int hi = pb + (6 + j * 3 + cc) * 8;   // rows for lanes 16..21
        if (j == 0) {   // gaussian 0 occupies K=0..8
          *(uint4*)&tiles[lo]     = make_uint4(q0, q1, q2, q3);
          *(uint4*)&tiles[lo + 4] = make_uint4(q4, 0, 0, 0);
          *(uint4*)&tiles[hi]     = make_uint4(0, 0, 0, 0);
          *(uint4*)&tiles[hi + 4] = make_uint4(0, 0, 0, 0);
        } else {        // gaussian 1 occupies K=16..24
          *(uint4*)&tiles[lo]     = make_uint4(0, 0, 0, 0);
          *(uint4*)&tiles[lo + 4] = make_uint4(0, 0, 0, 0);
          *(uint4*)&tiles[hi]     = make_uint4(q0, q1, q2, q3);
          *(uint4*)&tiles[hi + 4] = make_uint4(q4, 0, 0, 0);
        }
      }
    } else if (tid < 2 * CHUNK) {           // one thread per gaussian: params
      const int t = tid - CHUNK;
      const int n = ch * CHUNK + t;
      float mx = fminf(fmaxf(xyz[2 * n],     -1.f), 1.f);
      float my = fminf(fmaxf(xyz[2 * n + 1], -1.f), 1.f);
      float gx = mx * 268.8f + 256.f;   // (mu*1.05 + 1) * 0.5 * 512
      float gy = my * 268.8f + 256.f;
      float r  = rot[n]; r = r - floorf(r);
      float cs = __builtin_amdgcn_cosf(r);  // hw cos takes revolutions
      float sn = __builtin_amdgcn_sinf(r);
      float s0 = fminf(fmaxf(scal[2 * n],     0.f), 1.f) * S_SPAN + S_MIN;
      float s1 = fminf(fmaxf(scal[2 * n + 1], 0.f), 1.f) * S_SPAN + S_MIN;
      float op = fminf(fmaxf(opac[n], 0.f), 1.f);
      *(float4*)&params[t * 8]     = make_float4(gx, gy, cs, sn);
      *(float4*)&params[t * 8 + 4] = make_float4(s0, s1, op, 0.f);
    }
    __syncthreads();

    // software pipeline: async-issue next chunk's raw features now
    if (ch + 1 < NCHUNK) {
      unsigned long long sb = (unsigned long long)feat + (unsigned long long)(ch + 1) * RAWB;
      #pragma unroll
      for (int i = 0; i < 4; ++i) {
        int idx = i * 128 + tid;
        if (idx < NB128) async_b128(ldsRaw + idx * 16, (unsigned)(idx * 16), sb);
      }
    }

    // ---- compute: 16 rays x 2 gaussians per WMMA ----
    #pragma unroll 2
    for (int g = 0; g < PAIRS; ++g) {
      const int gl = (g << 1) | half;       // this lane's gaussian in chunk
      float4 p0 = *(const float4*)&params[gl * 8];
      float4 p1 = *(const float4*)&params[gl * 8 + 4];

      float vx = px - p0.x, vy = py - p0.y;
      float rx = p0.z * vx - p0.w * vy;
      float ry = p0.w * vx + p0.z * vy;
      float t0 = p1.x * rx, t1 = p1.y * ry;
      float d2 = t0 * t0 + t1 * t1;
      float w  = (d2 < 25.f) ? __builtin_amdgcn_exp2f(d2 * -L2E) * p1.z : 0.f;

      float vn  = __builtin_amdgcn_sqrtf(vx * vx + vy * vy);
      float inv = __builtin_amdgcn_rcpf(1e-10f + vn);
      // sin(theta)=rnx, cos(theta)=rny; Chebyshev-style angle addition
      float s1v = rx * inv,               c1v = ry * inv;
      float s2v = 2.f * s1v * c1v,        c2v = c1v * c1v - s1v * s1v;
      float s3v = s2v * c1v + c2v * s1v,  c3v = c2v * c1v - s2v * s1v;
      float s4v = 2.f * s2v * c2v,        c4v = c2v * c2v - s2v * s2v;

      v16h bb = {};                       // B operand: this lane's K-block
      bb[0] = (_Float16)1.0f;
      bb[1] = (_Float16)s1v; bb[2] = (_Float16)c1v;
      bb[3] = (_Float16)s2v; bb[4] = (_Float16)c2v;
      bb[5] = (_Float16)s3v; bb[6] = (_Float16)c3v;
      bb[7] = (_Float16)s4v; bb[8] = (_Float16)c4v;

      unsigned aoff = (col < 6) ? ((unsigned)(g * 96) + slotOff) : zOff;
      v16h av = *(const v16h*)&tiles[aoff];

      v8f cz = {};
      v8f d = __builtin_amdgcn_wmma_f32_16x16x32_f16(
          false, av, false, bb, (short)0, cz, false, false);

      // weight of the other half's gaussian: SWAPX16 swizzle (xor 0x10)
      float wo = __builtin_bit_cast(
          float, __builtin_amdgcn_ds_swizzle(__builtin_bit_cast(int, w), 0x401F));

      // rows 0..2 = gaussian(half=0) channels, rows 3..5 = gaussian(half=1)
      acc0 += w * sigm(d[0]) + wo * sigm(d[3]);
      acc1 += w * sigm(d[1]) + wo * sigm(d[4]);
      acc2 += w * sigm(d[2]) + wo * sigm(d[5]);
    }
    __syncthreads();   // tiles/params reused next chunk
  }

  if (lane < 16) {     // low half holds valid D columns / accumulators
    out[3 * ray]     = acc0;
    out[3 * ray + 1] = acc1;
    out[3 * ray + 2] = acc2;
  }
}

extern "C" void kernel_launch(void* const* d_in, const int* in_sizes, int n_in,
                              void* d_out, int out_size, void* d_ws, size_t ws_size,
                              hipStream_t stream) {
  (void)in_sizes; (void)n_in; (void)d_ws; (void)ws_size; (void)out_size;
  const float* xb   = (const float*)d_in[0];
  const float* xyz  = (const float*)d_in[1];
  const float* feat = (const float*)d_in[2];
  const float* opac = (const float*)d_in[3];
  const float* scal = (const float*)d_in[4];
  const float* rot  = (const float*)d_in[5];
  float* out = (float*)d_out;
  // 64 rays per block (4 waves x 16 D-columns), 8192/64 = 128 blocks
  splat2d_kernel<<<dim3(NB / 64), dim3(128), 0, stream>>>(
      xb, xyz, feat, opac, scal, rot, out);
}
